// CrossFeatureFusion_24283745091836
// MI455X (gfx1250) — matrix operand order repaired and checked by reading, not compiled, not run on hardware
//
#include <hip/hip_runtime.h>

// ---------------------------------------------------------------------------
// CrossFeatureFusion on MI455X (gfx1250), wave32 + WMMA bf16.
// out[i] = x[i] + sum_{j!=i} (x[j] @ W[i,j]^T + b[i,j])
// M=4, N=65536, D=256.
// Precision: fp32 emulated via bf16 hi/lo split (3 WMMA products per step).
// ---------------------------------------------------------------------------

typedef __attribute__((ext_vector_type(16))) __bf16 v16bf;
typedef __attribute__((ext_vector_type(8)))  __bf16 v8bf;
typedef __attribute__((ext_vector_type(4)))  __bf16 v4bf;
typedef __attribute__((ext_vector_type(8)))  float  v8f;

#define MM 4
#define DD 256
#define NB 32      // N-rows per workgroup
#define NT 2       // 16-row n-tiles per workgroup

// --------------------------- W -> bf16 hi/lo -------------------------------
__global__ __launch_bounds__(256)
void wconv_kernel(const float* __restrict__ W,
                  __bf16* __restrict__ Whi, __bf16* __restrict__ Wlo) {
    const int idx = (blockIdx.x * 256 + threadIdx.x) * 4;
    const float4 w = *(const float4*)(W + idx);
    float vs[4] = {w.x, w.y, w.z, w.w};
    v4bf hv, lv;
#pragma unroll
    for (int q = 0; q < 4; ++q) {
        __bf16 h = (__bf16)vs[q];
        hv[q] = h;
        lv[q] = (__bf16)(vs[q] - (float)h);
    }
    *(v4bf*)(Whi + idx) = hv;
    *(v4bf*)(Wlo + idx) = lv;
}

// ------------------------------ main kernel --------------------------------
__global__ __launch_bounds__(256)
void fusion_wmma_kernel(const float* __restrict__ x,
                        const __bf16* __restrict__ Whi,
                        const __bf16* __restrict__ Wlo,
                        const float* __restrict__ b,
                        float* __restrict__ out, int N) {
    __shared__ __bf16 sh_hi[MM * NB * DD];   // 64 KB
    __shared__ __bf16 sh_lo[MM * NB * DD];   // 64 KB

    const int t  = threadIdx.x;
    const int n0 = blockIdx.x * NB;

    // Stage 1: cooperatively load x tile (all 4 j's) and split into bf16 hi/lo.
    // Flat LDS layout [j][row][f] coincides with the flat load index.
    for (int v4 = t; v4 < (MM * NB * DD) / 4; v4 += 256) {
        const int e0 = v4 * 4;
        const int j  = e0 >> 13;           // / (NB*DD)
        const int r  = (e0 >> 8) & (NB - 1);
        const int f  = e0 & (DD - 1);
        const float4 xv = *(const float4*)(x + (((size_t)(j * N + n0 + r)) << 8) + f);
        float vs[4] = {xv.x, xv.y, xv.z, xv.w};
        v4bf hv, lv;
#pragma unroll
        for (int q = 0; q < 4; ++q) {
            __bf16 h = (__bf16)vs[q];
            hv[q] = h;
            lv[q] = (__bf16)(vs[q] - (float)h);
        }
        *(v4bf*)(&sh_hi[e0]) = hv;
        *(v4bf*)(&sh_lo[e0]) = lv;
    }
    __syncthreads();

    // Stage 2: WMMA. Wave w -> (i = w&3, e-half = w>>2).
    const int wave = t >> 5;
    const int lane = t & 31;
    const int i    = wave & 3;
    const int eh   = wave >> 2;       // 0 or 1 (columns 0..127 / 128..255)
    const int l16  = lane & 15;
    const int lhi  = lane >> 4;       // 0 or 1

    v8f acc[NT][8];
#pragma unroll
    for (int nt = 0; nt < NT; ++nt)
#pragma unroll
        for (int p = 0; p < 8; ++p)
            acc[nt][p] = v8f{};

    const int ebase = eh * 128;
    // B-fragment per-lane base into W (element index), without the (i,j) term:
    // B(K,e) = W[e][f]; lane holds column e = ebase + p*16 + l16,
    // K-run f = k*32 + lhi*16 + 0..15 (contiguous 32B).
    const int wlane = ((ebase + l16) << 8) + (lhi << 4);

#pragma unroll 1
    for (int jj = 1; jj < MM; ++jj) {
        const int j   = (i + jj) & 3;
        const int wjb = (((i << 2) + j) << 16);         // (i*4+j)*D*D
        const __bf16* whp = Whi + wjb + wlane;
        const __bf16* wlp = Wlo + wjb + wlane;
        // A-fragment per-lane base into LDS: row = l16 (+nt*16), chunks at
        // kb = lhi*16 and kb+8 within each 32-wide K-step.
        const int abase = ((j * NB + l16) * DD) + (lhi << 4);

#pragma unroll 1
        for (int k = 0; k < 8; ++k) {
            const int kb = k * 32;
            v16bf ah[NT], al[NT];
#pragma unroll
            for (int nt = 0; nt < NT; ++nt) {
                const int sa = abase + nt * (16 * DD) + kb;
                v8bf h0 = *(const v8bf*)(&sh_hi[sa]);
                v8bf h1 = *(const v8bf*)(&sh_hi[sa + 8]);
                v8bf l0 = *(const v8bf*)(&sh_lo[sa]);
                v8bf l1 = *(const v8bf*)(&sh_lo[sa + 8]);
                ah[nt] = __builtin_shufflevector(h0, h1, 0,1,2,3,4,5,6,7,8,9,10,11,12,13,14,15);
                al[nt] = __builtin_shufflevector(l0, l1, 0,1,2,3,4,5,6,7,8,9,10,11,12,13,14,15);
            }
#pragma unroll
            for (int p = 0; p < 8; ++p) {
                const int wo = (p << 12) + kb;          // p*16*D + k*32
                v16bf bh = *(const v16bf*)(whp + wo);
                v16bf bl = *(const v16bf*)(wlp + wo);
#pragma unroll
                for (int nt = 0; nt < NT; ++nt) {
                    acc[nt][p] = __builtin_amdgcn_wmma_f32_16x16x32_bf16(
                        false, ah[nt], false, bh, (short)0, acc[nt][p], false, false);
                    acc[nt][p] = __builtin_amdgcn_wmma_f32_16x16x32_bf16(
                        false, ah[nt], false, bl, (short)0, acc[nt][p], false, false);
                    acc[nt][p] = __builtin_amdgcn_wmma_f32_16x16x32_bf16(
                        false, al[nt], false, bh, (short)0, acc[nt][p], false, false);
                }
            }
        }
    }

    // Stage 3: epilogue. C/D layout: VGPR v -> row v + 8*(lane/16), col lane%16.
#pragma unroll 1
    for (int p = 0; p < 8; ++p) {
        const int e = ebase + p * 16 + l16;
        float bs = 0.0f;
#pragma unroll
        for (int jj = 1; jj < MM; ++jj) {
            const int j = (i + jj) & 3;
            bs += b[((((i << 2) + j)) << 8) + e];
        }
#pragma unroll
        for (int nt = 0; nt < NT; ++nt) {
            const int nrow = n0 + nt * 16 + (lhi << 3);
#pragma unroll
            for (int v = 0; v < 8; ++v) {
                const size_t oidx = (((size_t)(i * N + nrow + v)) << 8) + e;
                out[oidx] = x[oidx] + acc[nt][p][v] + bs;
            }
        }
    }
}

// ------------------- fallback (if workspace too small) ---------------------
__global__ __launch_bounds__(256)
void fusion_naive_kernel(const float* __restrict__ x, const float* __restrict__ W,
                         const float* __restrict__ b, float* __restrict__ out, int N) {
    const size_t gid = (size_t)blockIdx.x * 256 + threadIdx.x;
    const int e = (int)(gid & 255);
    const size_t in_ = gid >> 8;
    const int n = (int)(in_ % (size_t)N);
    const int i = (int)(in_ / (size_t)N);
    float s = x[gid];
    for (int jj = 1; jj < MM; ++jj) {
        const int j = (i + jj) & 3;
        const float* wr = W + ((size_t)((((i << 2) + j) << 8) + e) << 8);
        const float* xr = x + (((size_t)(j * N + n)) << 8);
        float a = b[((((i << 2) + j)) << 8) + e];
        for (int f = 0; f < DD; ++f) a = fmaf(wr[f], xr[f], a);
        s += a;
    }
    out[gid] = s;
}

// ------------------------------- launcher ----------------------------------
extern "C" void kernel_launch(void* const* d_in, const int* in_sizes, int n_in,
                              void* d_out, int out_size, void* d_ws, size_t ws_size,
                              hipStream_t stream) {
    const float* x = (const float*)d_in[0];
    const float* W = (const float*)d_in[1];
    const float* b = (const float*)d_in[2];
    float* out = (float*)d_out;

    const int N = in_sizes[0] / (MM * DD);         // 65536
    const size_t wElems = (size_t)MM * MM * DD * DD;  // 1,048,576
    const size_t wsNeed = wElems * 2 * sizeof(__bf16); // 4 MB (hi + lo)

    if (ws_size >= wsNeed) {
        __bf16* Whi = (__bf16*)d_ws;
        __bf16* Wlo = Whi + wElems;
        wconv_kernel<<<(int)(wElems / 1024), 256, 0, stream>>>(W, Whi, Wlo);
        fusion_wmma_kernel<<<N / NB, 256, 0, stream>>>(x, Whi, Wlo, b, out, N);
    } else {
        const size_t total = (size_t)MM * N * DD;
        fusion_naive_kernel<<<(int)(total / 256), 256, 0, stream>>>(x, W, b, out, N);
    }
}
